// MDRNN_558345749081
// MI455X (gfx1250) — matrix operand order repaired and checked by reading, not compiled, not run on hardware
//
#include <hip/hip_runtime.h>
#include <stdint.h>

typedef __attribute__((ext_vector_type(16))) __bf16 v16bf;
typedef __attribute__((ext_vector_type(8)))  __bf16 v8bf;
typedef __attribute__((ext_vector_type(8)))  float  v8f;

#define SGRID   29
#define BATCH   256
#define HDIM    256
#define HB_COLS 30
#define HB_CELL (BATCH * HDIM)                 // 65536 bf16 per (par,axis,col)
#define WPACK_ELEMS (4 * 17 * 48 * 32 * 16)    // 1,671,168 bf16
#define HBUF_ELEMS  (2 * 4 * HB_COLS * BATCH * HDIM)  // 15,728,640 bf16

__device__ __forceinline__ __bf16 f2bf(float f) {
    unsigned int u = __builtin_bit_cast(unsigned int, f);
    unsigned int r = u + 0x7FFFu + ((u >> 16) & 1u);   // round-to-nearest-even
    unsigned short h = (unsigned short)(r >> 16);
    return __builtin_bit_cast(__bf16, h);
}
__device__ __forceinline__ float bf2f(__bf16 b) {
    unsigned short h = __builtin_bit_cast(unsigned short, b);
    unsigned int u = ((unsigned int)h) << 16;
    return __builtin_bit_cast(float, u);
}

// ---------------------------------------------------------------- zero scratch
__global__ void zero_kernel(uint4* __restrict__ p, unsigned int n16) {
    unsigned int i = blockIdx.x * blockDim.x + threadIdx.x;
    unsigned int stride = gridDim.x * blockDim.x;
    uint4 z; z.x = z.y = z.z = z.w = 0u;
    for (; i < n16; i += stride) p[i] = z;
}

// ------------------------------------------------------- pack fused weights
// Wfull (per axis) = rows [Uh(256) ; Uh2(256) ; Wx(16) ; zeros(16)] x 768 cols.
// Tile (kt 0..16, nt 0..47) stored as 32 lanes x 16 bf16; lane L holds column
// nt*16 + (L&15), K = kt*32 + (e<8 ? (L>>4)*8 + e : 16 + (L>>4)*8 + e-8)  (B-fragment layout, wave32).
__global__ void pack_w_kernel(const float* __restrict__ Wx,
                              const float* __restrict__ Uh,
                              const float* __restrict__ Uh2,
                              __bf16* __restrict__ Wpack) {
    int idx = blockIdx.x * blockDim.x + threadIdx.x;
    if (idx >= WPACK_ELEMS) return;
    int e    = idx & 15;
    int t1   = idx >> 4;  int lane = t1 & 31;
    int t2   = t1 >> 5;   int nt   = t2 % 48;
    int t3   = t2 / 48;   int kt   = t3 % 17;
    int a    = t3 / 17;
    int ln = lane & 15, hi = lane >> 4;
    int k = kt * 32 + ((e < 8) ? (hi * 8 + e) : (16 + hi * 8 + (e - 8)));
    int n = nt * 16 + ln;
    float v = 0.f;
    if      (k < 256) v = Uh [(a * 256 +  k       ) * 768 + n];
    else if (k < 512) v = Uh2[(a * 256 + (k - 256)) * 768 + n];
    else if (k < 528) v = Wx [(a * 16  + (k - 512)) * 768 + n];
    Wpack[idx] = f2bf(v);
}

// ------------------------------------------------------------- wavefront cell
// One launch per anti-diagonal d; deps (y-1,x) and (y,x-1) are both exactly
// one diagonal back, so a ping-pong (parity, column) state buffer needs no
// copies. Block = 128 threads = 4 waves; each wave owns a 16-row x 768-col
// stripe of one cell's gate matrix (48 WMMA accum frags) plus 16 frags for
// the patch-only n-gate term (xn), so the r/z/n mixing — which needs columns
// j, j+256, j+512 — lands in the same lane & slot of different accumulators:
// the whole epilogue is per-lane register arithmetic.
__global__ __launch_bounds__(128, 1)
void cell_kernel(const float*  __restrict__ xin,
                 const float*  __restrict__ bias,
                 const __bf16* __restrict__ Wpack,
                 __bf16*       __restrict__ hbuf,
                 int d)
{
    int a  = blockIdx.y;
    int Ny = (a & 1) ? 28 : 29;                // bwd index list has length S-1
    int Nx = (a & 2) ? 28 : 29;
    int nd = Ny + Nx - 1;
    if (d >= nd) return;
    int ylo = d - (Nx - 1); if (ylo < 0) ylo = 0;
    int yhi = (d < Ny - 1) ? d : (Ny - 1);
    int y = ylo + (int)blockIdx.z;
    if (y > yhi) return;
    int xc   = d - y;
    int par  = d & 1;
    int prev = par ^ 1;

    int wave = threadIdx.x >> 5;
    int lane = threadIdx.x & 31;
    int ln = lane & 15, hi = lane >> 4;
    int r0 = (int)blockIdx.x * 64 + wave * 16;

    const __bf16* hA = hbuf + (size_t)((prev * 4 + a) * HB_COLS + (xc + 1)) * HB_CELL; // h(y-1,x)
    const __bf16* hL = hbuf + (size_t)((prev * 4 + a) * HB_COLS +  xc     ) * HB_CELL; // h(y,x-1); col0 = zero guard

    v8f zero8 = {0.f, 0.f, 0.f, 0.f, 0.f, 0.f, 0.f, 0.f};
    v8f acc[48];
    v8f accX[16];
#pragma unroll
    for (int i = 0; i < 48; ++i) acc[i] = zero8;
#pragma unroll
    for (int i = 0; i < 16; ++i) accX[i] = zero8;

    int arow = r0 + ln;
    int k0   = hi * 8;

    // K = 512 hidden-state part: kt 0..7 -> h_above, kt 8..15 -> h_left.
    // A-fragment = two contiguous b128 loads per lane from row-major bf16 h.
    for (int kt = 0; kt < 16; ++kt) {
        const __bf16* src = ((kt < 8) ? hA : hL) + arow * HDIM + (kt & 7) * 32 + k0;
        v8bf alo = *(const v8bf*)(src);
        v8bf ahi = *(const v8bf*)(src + 16);
        v16bf af = __builtin_shufflevector(alo, ahi, 0,1,2,3,4,5,6,7,8,9,10,11,12,13,14,15);
        const __bf16* bp = Wpack + (size_t)(a * 17 + kt) * 48 * 512 + lane * 16;
#pragma unroll
        for (int nt = 0; nt < 48; ++nt) {
            v16bf bf = *(const v16bf*)(bp + nt * 512);
            acc[nt] = __builtin_amdgcn_wmma_f32_16x16x32_bf16(
                false, af, false, bf, (short)0, acc[nt], false, false);
        }
    }

    // K-step 16: 4x4 patch (16 values, zero-padded to 32), gathered from x.
    {
        int Y = (a & 1) ? (27 - y)  : y;
        int X = (a & 2) ? (27 - xc) : xc;
        int dy0 = k0 >> 2;                    // hi=0 -> patch rows 0,1 ; hi=1 -> rows 2,3
        const float* px = xin + (size_t)arow * 1024 + (Y + dy0) * 32 + X;
        v16bf af;
#pragma unroll
        for (int i = 0; i < 4; ++i) af[i]     = f2bf(px[i]);
#pragma unroll
        for (int i = 0; i < 4; ++i) af[4 + i] = f2bf(px[32 + i]);
#pragma unroll
        for (int i = 8; i < 16; ++i) af[i]    = f2bf(0.f);
        const __bf16* bp = Wpack + (size_t)(a * 17 + 16) * 48 * 512 + lane * 16;
#pragma unroll
        for (int nt = 0; nt < 48; ++nt) {
            v16bf bf = *(const v16bf*)(bp + nt * 512);
            acc[nt] = __builtin_amdgcn_wmma_f32_16x16x32_bf16(
                false, af, false, bf, (short)0, acc[nt], false, false);
        }
#pragma unroll
        for (int nt = 0; nt < 16; ++nt) {     // xn (patch-only, n-gate cols)
            v16bf bf = *(const v16bf*)(bp + (32 + nt) * 512);
            accX[nt] = __builtin_amdgcn_wmma_f32_16x16x32_bf16(
                false, af, false, bf, (short)0, accX[nt], false, false);
        }
    }

    // Gate epilogue. acc[nt] = gr, acc[nt+16] = gz, acc[nt+32] = xn+hn+h2n,
    // accX[nt] = xn (all bias-free); biases cancel so hn+h2n = acc[nt+32]-accX.
    __bf16* hOut = hbuf + (size_t)((par * 4 + a) * HB_COLS + (xc + 1)) * HB_CELL;
    const float* ba = bias + a * 768;
#pragma unroll
    for (int nt = 0; nt < 16; ++nt) {
        int j = nt * 16 + ln;
        float bR = ba[j], bZ = ba[j + 256], bN = ba[j + 512];
#pragma unroll
        for (int v = 0; v < 8; ++v) {
            int row = r0 + v + 8 * hi;
            float r  = 1.f / (1.f + __expf(-(acc[nt][v]      + bR)));
            float z  = 1.f / (1.f + __expf(-(acc[nt + 16][v] + bZ)));
            float xn = accX[nt][v] + bN;
            float hsum = acc[nt + 32][v] - accX[nt][v];     // hn + h2n
            float n  = tanhf(xn + r * hsum);
            float ha = bf2f(hA[row * HDIM + j]);
            float hl = bf2f(hL[row * HDIM + j]);
            float hv = (1.f - z) * n + z * 0.5f * (ha + hl);
            hOut[row * HDIM + j] = f2bf(hv);
        }
    }
}

// ------------------------------------------------------------ output head
__global__ void out_kernel(const __bf16* __restrict__ hbuf,
                           const float*  __restrict__ W_out,
                           const float*  __restrict__ b_out,
                           float*        __restrict__ out)
{
    int b = threadIdx.x;                      // 256 threads, one per batch row
    float s[10];
#pragma unroll
    for (int o = 0; o < 10; ++o) s[o] = b_out[o];
    const int parc[4] = {0, 1, 1, 0};         // (nDiag-1)&1 per axis
    const int colc[4] = {29, 29, 28, 28};     // final column index (= Nx)
    for (int a = 0; a < 4; ++a) {
        const __bf16* hf = hbuf + (size_t)((parc[a] * 4 + a) * HB_COLS + colc[a]) * HB_CELL
                         + (size_t)b * HDIM;
        for (int j = 0; j < 256; ++j) {
            float hv = bf2f(hf[j]);
            const float* w = W_out + (a * 256 + j) * 10;
#pragma unroll
            for (int o = 0; o < 10; ++o) s[o] += hv * w[o];
        }
    }
    float m = s[0];
#pragma unroll
    for (int o = 1; o < 10; ++o) m = fmaxf(m, s[o]);
    float se = 0.f;
#pragma unroll
    for (int o = 0; o < 10; ++o) se += __expf(s[o] - m);
    float lse = m + __logf(se);
#pragma unroll
    for (int o = 0; o < 10; ++o) out[b * 10 + o] = s[o] - lse;
}

// ---------------------------------------------------------------- launcher
extern "C" void kernel_launch(void* const* d_in, const int* in_sizes, int n_in,
                              void* d_out, int out_size, void* d_ws, size_t ws_size,
                              hipStream_t stream) {
    const float* x    = (const float*)d_in[0];
    const float* Wx   = (const float*)d_in[1];
    const float* Uh   = (const float*)d_in[2];
    const float* Uh2  = (const float*)d_in[3];
    const float* bias = (const float*)d_in[4];
    const float* Wout = (const float*)d_in[5];
    const float* bout = (const float*)d_in[6];

    char* ws = (char*)d_ws;
    __bf16* Wpack = (__bf16*)ws;                         // 3.34 MB
    __bf16* hbuf  = (__bf16*)(ws + (4u << 20));          // 31.5 MB, 4MB-aligned

    unsigned int n16 = (unsigned int)((size_t)HBUF_ELEMS * 2 / 16);
    zero_kernel<<<512, 256, 0, stream>>>((uint4*)hbuf, n16);
    pack_w_kernel<<<(WPACK_ELEMS + 255) / 256, 256, 0, stream>>>(Wx, Uh, Uh2, Wpack);
    for (int d = 0; d < 57; ++d)
        cell_kernel<<<dim3(4, 4, 29), 128, 0, stream>>>(x, bias, Wpack, hbuf, d);
    out_kernel<<<1, 256, 0, stream>>>(hbuf, Wout, bout, (float*)d_out);
}
